// VQVAE_24017457119613
// MI455X (gfx1250) — matrix operand order repaired and checked by reading, not compiled, not run on hardware
//
#include <hip/hip_runtime.h>
#include <hip/hip_bf16.h>

// MI455X / gfx1250 VQ-VAE forward. 64-channel (de)convs + VQ distances run on
// v_wmma_f32_16x16x32_bf16 (bf16 in, f32 acc). Activation tiles are staged to
// LDS with global_load_async_to_lds_b32 (ASYNCcnt path). wave32, 16x16 tiles.

typedef __attribute__((ext_vector_type(16))) __bf16 v16bf;
typedef __attribute__((ext_vector_type(8)))  float  v8f;

__device__ __forceinline__ __bf16 f2bf(float f) {
  unsigned u = __float_as_uint(f);
  u = (u + 0x7FFFu + ((u >> 16) & 1u)) >> 16;   // round-to-nearest-even
  unsigned short s = (unsigned short)u;
  return __builtin_bit_cast(__bf16, s);
}
__device__ __forceinline__ float bf2f(__bf16 h) {
  unsigned short s = __builtin_bit_cast(unsigned short, h);
  return __uint_as_float(((unsigned)s) << 16);
}

#define WMMA_BF16(A, B, C) \
  __builtin_amdgcn_wmma_f32_16x16x32_bf16(false, (A), false, (B), (short)0, (C), false, false)

// One dword global -> LDS via the gfx1250 async path (no VGPR data round-trip).
__device__ __forceinline__ void async_g2l_b32(const void* gptr, void* lptr) {
  unsigned long long ga = (unsigned long long)(uintptr_t)gptr;
  unsigned la = (unsigned)(uintptr_t)lptr;   // generic LDS addr low 32 = wave LDS offset
  asm volatile("global_load_async_to_lds_b32 %0, %1, off" :: "v"(la), "v"(ga) : "memory");
}
__device__ __forceinline__ void wait_async0() {
  asm volatile("s_wait_asynccnt 0x0" ::: "memory");
}

// ---------------- weight / codebook prep ----------------

__global__ void k_prep_cast(const float* __restrict__ src, __bf16* __restrict__ dst, int n) {
  int i = blockIdx.x * blockDim.x + threadIdx.x;
  if (i < n) dst[i] = f2bf(src[i]);
}

// dec_w: (5, ci=64, co=64, k=4). Even phase taps k={1,3}, odd phase taps k={2,0}.
// wpe/wpo layout: [l][co][ci*2 + j], j=0 -> tap at t', j=1 -> tap at t'-1 / t'+1.
__global__ void k_prep_decw(const float* __restrict__ dw, __bf16* __restrict__ wpe,
                            __bf16* __restrict__ wpo) {
  int i = blockIdx.x * blockDim.x + threadIdx.x;   // 5*64*128 = 40960
  if (i >= 5 * 64 * 128) return;
  int j  = i & 1;
  int ci = (i >> 1) & 63;
  int co = (i >> 7) & 63;
  int l  = i >> 13;
  const float* w = dw + (size_t)l * 16384 + ci * 256 + co * 4;
  wpe[i] = f2bf(w[j ? 3 : 1]);
  wpo[i] = f2bf(w[j ? 0 : 2]);
}

__global__ void k_prep_cb(const float* __restrict__ cb, __bf16* __restrict__ cbb,
                          float* __restrict__ cn) {
  int c = blockIdx.x * blockDim.x + threadIdx.x;
  if (c >= 1024) return;
  float s = 0.f;
  for (int d = 0; d < 64; ++d) {
    float v = cb[c * 64 + d];
    s += v * v;
    cbb[c * 64 + d] = f2bf(v);
  }
  cn[c] = s;
}

// ---------------- encoder layer 0: conv 1->64 (VALU) ----------------

__global__ void k_enc0(const float* __restrict__ x, const float* __restrict__ w0,
                       const float* __restrict__ b0, __bf16* __restrict__ out,
                       int Lin, int Lout) {
  int gid = blockIdx.x * blockDim.x + threadIdx.x;   // 16*64*Lout
  int t  = gid % Lout;
  int co = (gid / Lout) & 63;
  int b  = gid / (Lout * 64);
  float acc = b0[co];
  int base = 2 * t - 1;
#pragma unroll
  for (int k = 0; k < 4; ++k) {
    int ti = base + k;
    float xv = (ti >= 0 && ti < Lin) ? x[(size_t)b * Lin + ti] : 0.f;
    acc += w0[co * 4 + k] * xv;
  }
  out[gid] = f2bf(fmaxf(acc, 0.f));
}

// ---------------- conv 64->64 k4 s2 p1 + relu (WMMA GEMM) ----------------
// block = 128 threads (4 waves); wave w -> co tile [16w,16w+16); blockIdx.x -> 16 t outputs.
// K dim = ci*4 + tap (256), 8 chunks of 32. LDS tile: 64 ci x 36 positions, async-staged,
// window starts at global t = 2*t0-2 (4B-aligned); p=0/35 unused, p=1/34 zeroed at edges.

__global__ void k_conv64(const __bf16* __restrict__ in, const __bf16* __restrict__ wp,
                         const float* __restrict__ bias, __bf16* __restrict__ out,
                         float* __restrict__ ze, int Lin, int Lout) {
  __shared__ __bf16 tile[64 * 36];
  int b  = blockIdx.y;
  int t0 = blockIdx.x * 16;
  const __bf16* inb = in + (size_t)b * 64 * Lin;
  // 64 rows * 18 dwords = 1152 dwords, 128 lanes * 9 iters
#pragma unroll
  for (int it = 0; it < 9; ++it) {
    int idx = it * 128 + (int)threadIdx.x;
    int ci = idx / 18, d = idx % 18;
    long long goff = ((long long)ci * Lin + 2 * t0 - 2) * 2 + d * 4;   // bytes
    async_g2l_b32((const char*)inb + goff, &tile[ci * 36 + d * 2]);
  }
  wait_async0();
  if (blockIdx.x == 0)                                   // t = -1 zero-pad
    for (int ci = threadIdx.x; ci < 64; ci += 128) tile[ci * 36 + 1] = f2bf(0.f);
  if (t0 == Lout - 16)                                   // t = Lin zero-pad
    for (int ci = threadIdx.x; ci < 64; ci += 128) tile[ci * 36 + 34] = f2bf(0.f);
  __syncthreads();

  int lane = threadIdx.x & 31, wv = threadIdx.x >> 5;
  int half = lane >> 4, nm = lane & 15;
  v8f acc = {};
#pragma unroll
  for (int c = 0; c < 8; ++c) {
    v16bf A, B;
#pragma unroll
    for (int i = 0; i < 16; ++i) {
      int ka = c * 32 + i + (i & 8) + half * 8;    // A: K per ISA 16-bit A layout
      A[i] = wp[(16 * wv + nm) * 256 + ka];
      int kb = i + (i & 8) + half * 8;             // B: K per ISA 16-bit B layout
      int ci = c * 8 + (kb >> 2), kt = kb & 3;
      B[i] = tile[ci * 36 + 2 * nm + kt + 1];      // x[ci, 2t + kt - 1]
    }
    acc = WMMA_BF16(A, B, acc);
  }

  __bf16* outb = out + (size_t)b * 64 * Lout;
#pragma unroll
  for (int r = 0; r < 8; ++r) {
    int co = 16 * wv + r + 8 * half;
    int t  = t0 + nm;
    float v = fmaxf(acc[r] + bias[co], 0.f);
    outb[(size_t)co * Lout + t] = f2bf(v);
    if (ze) ze[((size_t)b * Lout + t) * 64 + co] = v;   // token-major z_e (fp32 output)
  }
}

// ---------------- VQ: argmin ||z - c||^2 + gather (WMMA for z@c^T) ----------------
// block = 256 threads (8 waves) handles 16 tokens x 1024 codes; wave handles 8 code tiles.

__global__ void k_vq(const float* __restrict__ ze, const __bf16* __restrict__ cbb,
                     const float* __restrict__ cbf, const float* __restrict__ cnorm,
                     float* __restrict__ ids_out, __bf16* __restrict__ g) {
  __shared__ float znorm[16];
  __shared__ unsigned long long best[16];
  __shared__ int sid[16];
  int tok0 = blockIdx.x * 16;
  if (threadIdx.x < 16) {
    best[threadIdx.x] = ~0ull;
    const float* z = ze + (size_t)(tok0 + threadIdx.x) * 64;
    float s = 0.f;
    for (int d = 0; d < 64; ++d) s += z[d] * z[d];
    znorm[threadIdx.x] = s;
  }
  __syncthreads();

  int lane = threadIdx.x & 31, wv = threadIdx.x >> 5;
  int half = lane >> 4, nm = lane & 15;

  v16bf A0, A1;                                    // z tile, K chunks 0/1
#pragma unroll
  for (int i = 0; i < 16; ++i) {
    int ka = i + (i & 8) + half * 8;
    A0[i] = f2bf(ze[(size_t)(tok0 + nm) * 64 + ka]);
    A1[i] = f2bf(ze[(size_t)(tok0 + nm) * 64 + 32 + ka]);
  }

  unsigned long long bl[8];
#pragma unroll
  for (int r = 0; r < 8; ++r) bl[r] = ~0ull;

  for (int ct = wv; ct < 64; ct += 8) {
    v8f acc = {};
    v16bf B0, B1;
#pragma unroll
    for (int i = 0; i < 16; ++i) {
      int kb = i + (i & 8) + half * 8;
      B0[i] = cbb[(size_t)(ct * 16 + nm) * 64 + kb];
      B1[i] = cbb[(size_t)(ct * 16 + nm) * 64 + 32 + kb];
    }
    acc = WMMA_BF16(A0, B0, acc);
    acc = WMMA_BF16(A1, B1, acc);
    int code = ct * 16 + nm;
    float cn = cnorm[code];
#pragma unroll
    for (int r = 0; r < 8; ++r) {
      int m = r + 8 * half;
      float dist = znorm[m] - 2.f * acc[r] + cn;
      unsigned u = __float_as_uint(dist);
      u = (u & 0x80000000u) ? ~u : (u | 0x80000000u);          // order-preserving key
      unsigned long long key = ((unsigned long long)u << 32) | (unsigned)code;
      if (key < bl[r]) bl[r] = key;                            // ties -> lowest id
    }
  }
#pragma unroll
  for (int r = 0; r < 8; ++r)
    atomicMin(&best[r + 8 * half], bl[r]);                     // ds_min_u64
  __syncthreads();

  if (threadIdx.x < 16) {
    int id = (int)(best[threadIdx.x] & 0xFFFFFFFFull);
    sid[threadIdx.x] = id;
    ids_out[tok0 + threadIdx.x] = (float)id;
  }
  __syncthreads();

  int b = tok0 >> 10, tl0 = tok0 & 1023;
  for (int idx = threadIdx.x; idx < 16 * 64; idx += blockDim.x) {
    int tl = idx >> 6, d = idx & 63;
    g[((size_t)b * 64 + d) * 1024 + (tl0 + tl)] = f2bf(cbf[sid[tl] * 64 + d]);
  }
}

// ---------------- conv-transpose 64->64 k4 s2 p1 + relu (WMMA GEMM) ----------------
// Even (u=2t') / odd (u=2t'+1) phases; K dim = ci*2+j (128), 4 chunks. LDS tile:
// 64 ci x 20 positions, async-staged, window starts at t = t0-2; p=0/19 unused.

__global__ void k_deconv64(const __bf16* __restrict__ in, const __bf16* __restrict__ wpe,
                           const __bf16* __restrict__ wpo, const float* __restrict__ bias,
                           __bf16* __restrict__ out, int Lin) {
  __shared__ __bf16 tile[64 * 20];
  int b  = blockIdx.y;
  int t0 = blockIdx.x * 16;
  const __bf16* inb = in + (size_t)b * 64 * Lin;
  // 64 rows * 10 dwords = 640 dwords, 128 lanes * 5 iters
#pragma unroll
  for (int it = 0; it < 5; ++it) {
    int idx = it * 128 + (int)threadIdx.x;
    int ci = idx / 10, d = idx % 10;
    long long goff = ((long long)ci * Lin + t0 - 2) * 2 + d * 4;   // bytes
    async_g2l_b32((const char*)inb + goff, &tile[ci * 20 + d * 2]);
  }
  wait_async0();
  if (blockIdx.x == 0)                                   // t = -1 zero-pad
    for (int ci = threadIdx.x; ci < 64; ci += 128) tile[ci * 20 + 1] = f2bf(0.f);
  if (t0 == Lin - 16)                                    // t = Lin zero-pad
    for (int ci = threadIdx.x; ci < 64; ci += 128) tile[ci * 20 + 18] = f2bf(0.f);
  __syncthreads();

  int lane = threadIdx.x & 31, wv = threadIdx.x >> 5;
  int half = lane >> 4, nm = lane & 15;
  v8f accE = {}, accO = {};
#pragma unroll
  for (int c = 0; c < 4; ++c) {
    v16bf AE, AO, BE, BO;
#pragma unroll
    for (int i = 0; i < 16; ++i) {
      int ka = c * 32 + i + (i & 8) + half * 8;
      AE[i] = wpe[(16 * wv + nm) * 128 + ka];
      AO[i] = wpo[(16 * wv + nm) * 128 + ka];
      int kk = i + (i & 8) + half * 8;
      int ci = c * 16 + (kk >> 1), j = kk & 1;
      BE[i] = tile[ci * 20 + (j ? nm + 1 : nm + 2)];  // x[t'-1] / x[t']
      BO[i] = tile[ci * 20 + (j ? nm + 3 : nm + 2)];  // x[t'+1] / x[t']
    }
    accE = WMMA_BF16(AE, BE, accE);
    accO = WMMA_BF16(AO, BO, accO);
  }

  int Lout = 2 * Lin;
  __bf16* outb = out + (size_t)b * 64 * Lout;
#pragma unroll
  for (int r = 0; r < 8; ++r) {
    int co = 16 * wv + r + 8 * half;
    int tp = t0 + nm;
    float bv = bias[co];
    outb[(size_t)co * Lout + 2 * tp]     = f2bf(fmaxf(accE[r] + bv, 0.f));
    outb[(size_t)co * Lout + 2 * tp + 1] = f2bf(fmaxf(accO[r] + bv, 0.f));
  }
}

// ---------------- decoder last: conv-transpose 64->1, no relu (VALU) ----------------

__global__ void k_declast(const __bf16* __restrict__ in, const float* __restrict__ wl,
                          const float* __restrict__ blast, float* __restrict__ y, int Lin) {
  int gid = blockIdx.x * blockDim.x + threadIdx.x;   // 16 * 2*Lin
  int Lout = 2 * Lin;
  int u = gid % Lout, b = gid / Lout;
  const __bf16* inb = in + (size_t)b * 64 * Lin;
  int tp = u >> 1;
  bool even = !(u & 1);
  float acc = blast[0];
  for (int ci = 0; ci < 64; ++ci) {
    float x0 = bf2f(inb[(size_t)ci * Lin + tp]);
    if (even) {
      acc += wl[ci * 4 + 1] * x0;
      if (tp - 1 >= 0) acc += wl[ci * 4 + 3] * bf2f(inb[(size_t)ci * Lin + tp - 1]);
    } else {
      acc += wl[ci * 4 + 2] * x0;
      if (tp + 1 < Lin) acc += wl[ci * 4 + 0] * bf2f(inb[(size_t)ci * Lin + tp + 1]);
    }
  }
  y[gid] = acc;
}

// ---------------- host launch ----------------

extern "C" void kernel_launch(void* const* d_in, const int* in_sizes, int n_in,
                              void* d_out, int out_size, void* d_ws, size_t ws_size,
                              hipStream_t stream) {
  (void)in_sizes; (void)n_in; (void)out_size; (void)ws_size;
  const float* x   = (const float*)d_in[0];
  const float* ew0 = (const float*)d_in[1];
  const float* ew  = (const float*)d_in[2];
  const float* eb  = (const float*)d_in[3];
  const float* cb  = (const float*)d_in[4];
  const float* dw  = (const float*)d_in[5];
  const float* dwl = (const float*)d_in[6];
  const float* db  = (const float*)d_in[7];
  const float* dbl = (const float*)d_in[8];

  float* out  = (float*)d_out;
  float* y    = out;                         // (16,1,65536)
  float* ze   = out + 16 * 65536;            // (16384,64)
  float* idsf = out + 2 * 16 * 65536;        // (16384,) as float

  // workspace layout (~135 MB assumed); 256B guard bands around the ping-pong
  // buffers make the async-staging +/-4B window overreads safe.
  char* ws = (char*)d_ws;
  size_t o = 256;
  __bf16* bufA  = (__bf16*)(ws + o); o += ((size_t)64 << 20) + 256;
  __bf16* bufB  = (__bf16*)(ws + o); o += ((size_t)64 << 20) + 256;
  __bf16* encWp = (__bf16*)(ws + o);                         // 5*64*256
  __bf16* wpe   = encWp + 5 * 16384;                         // 5*64*128
  __bf16* wpo   = wpe + 5 * 8192;
  __bf16* cbb   = wpo + 5 * 8192;                            // 1024*64
  float*  cn    = (float*)(cbb + 1024 * 64);                 // 1024

  k_prep_cast<<<(81920 + 255) / 256, 256, 0, stream>>>(ew, encWp, 81920);
  k_prep_decw<<<(40960 + 255) / 256, 256, 0, stream>>>(dw, wpe, wpo);
  k_prep_cb<<<4, 256, 0, stream>>>(cb, cbb, cn);

  // encoder
  k_enc0<<<(16 * 64 * 32768) / 256, 256, 0, stream>>>(x, ew0, eb, bufA, 65536, 32768);
  __bf16* cur = bufA; __bf16* nxt = bufB;
  for (int l = 0; l < 5; ++l) {
    int Lin = 32768 >> l, Lout = Lin >> 1;
    float* zptr = (l == 4) ? ze : nullptr;
    k_conv64<<<dim3(Lout / 16, 16), 128, 0, stream>>>(cur, encWp + l * 16384,
                                                      eb + (l + 1) * 64, nxt, zptr, Lin, Lout);
    __bf16* t = cur; cur = nxt; nxt = t;
  }

  // vector quantize; writes decoder input g into the free buffer
  __bf16* g = nxt;
  k_vq<<<1024, 256, 0, stream>>>(ze, cbb, cb, cn, idsf, g);

  // decoder
  cur = g; nxt = (g == bufA) ? bufB : bufA;
  for (int l = 0; l < 5; ++l) {
    int Lin = 1024 << l;
    k_deconv64<<<dim3(Lin / 16, 16), 128, 0, stream>>>(cur, wpe + l * 8192, wpo + l * 8192,
                                                       db + l * 64, nxt, Lin);
    __bf16* t = cur; cur = nxt; nxt = t;
  }
  k_declast<<<(16 * 65536) / 256, 256, 0, stream>>>(cur, dwl, dbl, y, 32768);
}
